// Basic_RSNN_eprop_forward_88845693485085
// MI455X (gfx1250) — matrix operand order repaired and checked by reading, not compiled, not run on hardware
//
#include <hip/hip_runtime.h>
#include <hip/hip_bf16.h>

// ---------------------------------------------------------------------------
// e-prop RSNN step, MI455X (gfx1250), wave32 + WMMA f32<-f16 16x16x32.
// Per-timestep kernel chain (recurrence forces serialization); all weights,
// state and gradients are L2-resident; GEMMs are WMMA-tiled with register
// blocking; hidden-layer B panels staged via async global->LDS (ASYNCcnt).
// ---------------------------------------------------------------------------

#define B_    64
#define T_    256
#define NIN   512
#define NHID  1024
#define NOUT  128

#define TAU_   0.6f
#define TAUO_  0.6f
#define THR_   0.6f
#define GAMMA_ 0.3f
#define LR_    0.01f

typedef __attribute__((ext_vector_type(16))) _Float16 v16h;
typedef __attribute__((ext_vector_type(8)))  _Float16 v8h;
typedef __attribute__((ext_vector_type(8)))  float    v8f;
typedef __attribute__((ext_vector_type(4)))  int      v4i;

typedef __attribute__((address_space(1))) v4i gv4i;   // global (device)
typedef __attribute__((address_space(3))) v4i sv4i;   // LDS (shared)

// ---- async global -> LDS staging (CDNA5, tracked by ASYNCcnt) -------------

__device__ __forceinline__ void async_b128(const void* g, void* l) {
  // AS1 pointer value == flat value; AS3 value == low 32 bits of generic
  // LDS address (ISA 10.2).
  gv4i* gp = (gv4i*)(unsigned long long)(__SIZE_TYPE__)g;
  sv4i* lp = (sv4i*)(unsigned int)(__SIZE_TYPE__)l;
#if __has_builtin(__builtin_amdgcn_global_load_async_to_lds_b128)
  __builtin_amdgcn_global_load_async_to_lds_b128(gp, lp, 0, 0);
#else
  unsigned int laddr = (unsigned int)(__SIZE_TYPE__)l;
  asm volatile("global_load_async_to_lds_b128 %0, %1, off"
               :: "v"(laddr), "v"(g) : "memory");
#endif
}

__device__ __forceinline__ void wait_async0() {
#if __has_builtin(__builtin_amdgcn_s_wait_asynccnt)
  __builtin_amdgcn_s_wait_asynccnt(0);
#else
  asm volatile("s_wait_asynccnt 0x0" ::: "memory");
#endif
}

// ---- WMMA operand loaders matching CDNA5 16-bit layouts (CDNA5 ISA) ------

// A (16x32, row-major src [M][K], ld elems). lane<16: K 0-7 & 16-23;
// lane>=16: K 8-15 & 24-31 (same row M = lane&15).
__device__ __forceinline__ v16h load_a(const _Float16* src, int ld) {
  const int l  = threadIdx.x & 31;
  const int ml = l & 15;
  const int hi = l >> 4;
  const _Float16* p = src + ml * ld + hi * 8;
  v8h c0 = *(const v8h*)(p);
  v8h c1 = *(const v8h*)(p + 16);
  v16h a;
  #pragma unroll
  for (int i = 0; i < 8; ++i) { a[i] = c0[i]; a[8 + i] = c1[i]; }
  return a;
}

// B (32x16) from BT stored row-major [N][K] (ld elems). lane n<16 holds
// K 0-7 & 16-23 of column n; lane n+16 holds K 8-15 & 24-31.
__device__ __forceinline__ v16h load_b(const _Float16* bt, int ld) {
  const int l  = threadIdx.x & 31;
  const int nl = l & 15;
  const int hi = l >> 4;
  const _Float16* p = bt + nl * ld + hi * 8;
  v8h c0 = *(const v8h*)(p);
  v8h c1 = *(const v8h*)(p + 16);
  v16h b;
  #pragma unroll
  for (int i = 0; i < 8; ++i) { b[i] = c0[i]; b[8 + i] = c1[i]; }
  return b;
}

__device__ __forceinline__ v8f wmma(v16h a, v16h b, v8f c) {
  return __builtin_amdgcn_wmma_f32_16x16x32_f16(
      false, a, false, b, (short)0, c, false, false);
}

// One 16x16 output tile, K swept by 32.
__device__ __forceinline__ v8f wmma_tile(const _Float16* A, int lda,
                                         const _Float16* BT, int ldb,
                                         int K, v8f c) {
  for (int k0 = 0; k0 < K; k0 += 32)
    c = wmma(load_a(A + k0, lda), load_b(BT + k0, ldb), c);
  return c;
}

// ---------------------------- prep kernels ---------------------------------

__global__ void k_zero(float* p, int n) {
  int i = blockIdx.x * blockDim.x + threadIdx.x;
  if (i < n) p[i] = 0.f;
}

__global__ void k_cvt_win(const float* w_in, _Float16* winh) {
  int i = blockIdx.x * blockDim.x + threadIdx.x;       // NHID*NIN
  winh[i] = (_Float16)w_in[i];
}

__global__ void k_wrecT(const float* w_rec, _Float16* wrecT) {
  int i = blockIdx.x * blockDim.x + threadIdx.x;       // NHID*NHID, i = r*NHID+j
  int r = i >> 10, j = i & 1023;
  wrecT[i] = (_Float16)w_rec[j * NHID + r];
}

__global__ void k_cvt_wout(const float* w_out, _Float16* wouth, _Float16* woutT) {
  int i = blockIdx.x * blockDim.x + threadIdx.x;       // NOUT*NHID, i = o*NHID+r
  int o = i >> 10, r = i & 1023;
  float v = w_out[i];
  wouth[i] = (_Float16)v;
  woutT[r * NOUT + o] = (_Float16)v;
}

// ------------------------- per-timestep kernels ----------------------------

// x[:, t, :] -> f16 [B][NIN]
__global__ void k_xconv(const float* x, int t, _Float16* xth) {
  int i = blockIdx.x * blockDim.x + threadIdx.x;       // B*NIN
  int b = i >> 9, ii = i & 511;
  xth[i] = (_Float16)x[(b * T_ + t) * NIN + ii];
}

// tr_in (feature-major [i][b]) update: tr = TAU*tr + x_t
__global__ void k_trin(const float* x, int t, float* trin, _Float16* trinh) {
  int i = blockIdx.x * blockDim.x + threadIdx.x;       // NIN*B, i = ii*B + b
  int ii = i >> 6, b = i & 63;
  float v = TAU_ * trin[i] + x[(b * T_ + t) * NIN + ii];
  trin[i] = v;
  trinh[i] = (_Float16)v;
}

// hidden LIF: cur = x_t @ w_in^T + h_spk @ w_rec ; hard-reset membrane.
// grid = NHID/32 blocks (32 r columns each), 128 threads = 4 waves over b;
// each wave holds 2 accumulators (2 n-tiles) -> 2 WMMAs per global A fetch.
__global__ __launch_bounds__(128) void k_hidden(
    const _Float16* __restrict__ xth, const _Float16* __restrict__ winh,
    const _Float16* __restrict__ wrecT, const _Float16* __restrict__ hspk_prev,
    _Float16* __restrict__ hspk_next, float* __restrict__ hmem) {
  __shared__ __align__(16) _Float16 sB[32 * (NIN + NHID)];  // 96 KB
  const int n0 = blockIdx.x * 32;

  __builtin_prefetch(xth, 0, 3);  // pull activation panel toward the WGP

  // async-stage B panels (32 rows of winh + 32 rows of wrecT, contiguous)
  {
    const _Float16* g1 = winh + n0 * NIN;
    for (int i = threadIdx.x; i < 32 * NIN / 8; i += blockDim.x)
      async_b128(g1 + i * 8, sB + i * 8);
    const _Float16* g2 = wrecT + n0 * NHID;
    for (int i = threadIdx.x; i < 32 * NHID / 8; i += blockDim.x)
      async_b128(g2 + i * 8, sB + 32 * NIN + i * 8);
    wait_async0();
  }
  __syncthreads();

  const int m0 = (threadIdx.x >> 5) * 16;
  v8f c0 = {}, c1 = {};
  const _Float16* A1 = xth + m0 * NIN;
  for (int k0 = 0; k0 < NIN; k0 += 32) {
    v16h a  = load_a(A1 + k0, NIN);
    v16h b0 = load_b(sB + k0, NIN);
    v16h b1 = load_b(sB + 16 * NIN + k0, NIN);
    c0 = wmma(a, b0, c0);
    c1 = wmma(a, b1, c1);
  }
  const _Float16* A2 = hspk_prev + m0 * NHID;
  const _Float16* B2 = sB + 32 * NIN;
  for (int k0 = 0; k0 < NHID; k0 += 32) {
    v16h a  = load_a(A2 + k0, NHID);
    v16h b0 = load_b(B2 + k0, NHID);
    v16h b1 = load_b(B2 + 16 * NHID + k0, NHID);
    c0 = wmma(a, b0, c0);
    c1 = wmma(a, b1, c1);
  }

  const int l = threadIdx.x & 31, nl = l & 15, hi = l >> 4;
  #pragma unroll
  for (int v = 0; v < 8; ++v) {
    int b = m0 + v + 8 * hi;
    #pragma unroll
    for (int tn = 0; tn < 2; ++tn) {
      int r = n0 + tn * 16 + nl;
      int idx = b * NHID + r;
      float cur = tn ? c1[v] : c0[v];
      float prev = (float)hspk_prev[idx];               // 0/1, exact in f16
      float mem = TAU_ * hmem[idx] * (1.f - prev) + cur;
      float spk = (mem - THR_ >= 0.f) ? 1.f : 0.f;
      hmem[idx] = mem;
      hspk_next[idx] = (_Float16)spk;
    }
  }
}

// output LIF + err; writes o_spk to d_out[:, t, :], err in both layouts.
// grid = NOUT/16 = 8 blocks, 4 waves over b tiles.
__global__ __launch_bounds__(128) void k_out(
    const _Float16* __restrict__ hspk, const _Float16* __restrict__ wouth,
    const float* __restrict__ label, int t, float* __restrict__ omem,
    float* __restrict__ ospk, float* __restrict__ dout,
    _Float16* __restrict__ errh, _Float16* __restrict__ errT) {
  const int n0 = blockIdx.x * 16;
  const int m0 = (threadIdx.x >> 5) * 16;
  v8f c = {};
  c = wmma_tile(hspk + m0 * NHID, NHID, wouth + n0 * NHID, NHID, NHID, c);

  const int l = threadIdx.x & 31, nl = l & 15, hi = l >> 4;
  #pragma unroll
  for (int v = 0; v < 8; ++v) {
    int b = m0 + v + 8 * hi, o = n0 + nl;
    int idx = b * NOUT + o;
    float osp = ospk[idx];
    float mem = TAU_ * omem[idx] * (1.f - osp) + c[v];
    float spk = (mem - THR_ >= 0.f) ? 1.f : 0.f;
    omem[idx] = mem;
    ospk[idx] = spk;
    float err = spk - label[(b * T_ + t) * NOUT + o];
    dout[(b * T_ + t) * NOUT + o] = spk;
    errh[idx] = (_Float16)err;
    errT[o * B_ + b] = (_Float16)err;
  }
}

// learning signal Lh = (err @ w_out) * h_t, stored transposed [r][b];
// also exponential rec/out trace updates (feature-major).
// grid = NHID/16 = 64 blocks, 4 waves over b tiles.
__global__ __launch_bounds__(128) void k_ls(
    const _Float16* __restrict__ errh, const _Float16* __restrict__ woutT,
    const float* __restrict__ hmem, const _Float16* __restrict__ hspk,
    _Float16* __restrict__ lhT, float* __restrict__ trrec,
    _Float16* __restrict__ trrech, float* __restrict__ trout,
    _Float16* __restrict__ trouth) {
  const int n0 = blockIdx.x * 16;
  const int m0 = (threadIdx.x >> 5) * 16;
  v8f c = {};
  c = wmma_tile(errh + m0 * NOUT, NOUT, woutT + n0 * NOUT, NOUT, NOUT, c);

  const int l = threadIdx.x & 31, nl = l & 15, hi = l >> 4;
  #pragma unroll
  for (int v = 0; v < 8; ++v) {
    int b = m0 + v + 8 * hi, r = n0 + nl;
    float hm = hmem[b * NHID + r];
    float ht = GAMMA_ * fmaxf(0.f, 1.f - fabsf((hm - THR_) / THR_));
    lhT[r * B_ + b] = (_Float16)(c[v] * ht);
    float spk = (float)hspk[b * NHID + r];
    int ti = r * B_ + b;
    float tr = TAU_ * trrec[ti] + spk;
    trrec[ti] = tr; trrech[ti] = (_Float16)tr;
    float to = TAUO_ * trout[ti] + spk;
    trout[ti] = to; trouth[ti] = (_Float16)to;
  }
}

// rank-64 gradient accumulation, 2x2 register-tiled: each wave job covers a
// 32x32 output patch (8 WMMAs, 2 A-loads + 2 B-loads per k-step).
// Jobs: g_in 512 (32x16), g_rec 1024 (32x32), g_out 128 (4x32) = 1664.
__global__ __launch_bounds__(128) void k_grad(
    const _Float16* __restrict__ lhT, const _Float16* __restrict__ errT,
    const _Float16* __restrict__ trinh, const _Float16* __restrict__ trrech,
    const _Float16* __restrict__ trouth, float* __restrict__ gin,
    float* __restrict__ grec, float* __restrict__ gout) {
  const int wid = blockIdx.x * 4 + (threadIdx.x >> 5);
  const _Float16 *A, *BT;
  float* D;
  int m0, n0, ldd;
  if (wid < 512) {                        // g_in [NHID][NIN]
    int tm = wid >> 4, tn = wid & 15;
    m0 = tm * 32; n0 = tn * 32;
    A = lhT + m0 * B_; BT = trinh + n0 * B_; D = gin; ldd = NIN;
  } else if (wid < 1536) {                // g_rec [NHID][NHID]
    int id = wid - 512, tm = id >> 5, tn = id & 31;
    m0 = tm * 32; n0 = tn * 32;
    A = lhT + m0 * B_; BT = trrech + n0 * B_; D = grec; ldd = NHID;
  } else {                                // g_out [NOUT][NHID]
    int id = wid - 1536, tm = id >> 5, tn = id & 31;
    m0 = tm * 32; n0 = tn * 32;
    A = errT + m0 * B_; BT = trouth + n0 * B_; D = gout; ldd = NHID;
  }

  v8f c00 = {}, c01 = {}, c10 = {}, c11 = {};
  for (int k0 = 0; k0 < B_; k0 += 32) {   // K = 64 -> 2 k-steps, 8 WMMAs
    v16h a0 = load_a(A + k0, B_);
    v16h a1 = load_a(A + 16 * B_ + k0, B_);
    v16h b0 = load_b(BT + k0, B_);
    v16h b1 = load_b(BT + 16 * B_ + k0, B_);
    c00 = wmma(a0, b0, c00);
    c01 = wmma(a0, b1, c01);
    c10 = wmma(a1, b0, c10);
    c11 = wmma(a1, b1, c11);
  }

  const int l = threadIdx.x & 31, nl = l & 15, hi = l >> 4;
  #pragma unroll
  for (int v = 0; v < 8; ++v) {
    int m = m0 + v + 8 * hi;
    D[m * ldd + (n0 + nl)]             += LR_ * c00[v];
    D[m * ldd + (n0 + 16 + nl)]        += LR_ * c01[v];
    D[(m + 16) * ldd + (n0 + nl)]      += LR_ * c10[v];
    D[(m + 16) * ldd + (n0 + 16 + nl)] += LR_ * c11[v];
  }
}

// ------------------------------ host side ----------------------------------

extern "C" void kernel_launch(void* const* d_in, const int* in_sizes, int n_in,
                              void* d_out, int out_size, void* d_ws, size_t ws_size,
                              hipStream_t stream) {
  const float* x     = (const float*)d_in[0];
  const float* label = (const float*)d_in[1];
  const float* w_in  = (const float*)d_in[2];
  const float* w_rec = (const float*)d_in[3];
  const float* w_out = (const float*)d_in[4];
  float* dout = (float*)d_out;

  // workspace layout (bytes)
  char* ws = (char*)d_ws;
  _Float16* winh   = (_Float16*)(ws);                         // 1 MB
  _Float16* wrecT  = (_Float16*)(ws + (1u << 20));            // 2 MB
  _Float16* wouth  = (_Float16*)(ws + 3 * (1u << 20));        // 256 KB
  _Float16* woutT  = (_Float16*)(ws + 3 * (1u << 20) + 262144);
  _Float16* xth    = (_Float16*)(ws + 3 * (1u << 20) + 524288); // 64 KB
  char* st = ws + 3 * (1u << 20) + 524288 + 65536;            // state span
  float*    hmem   = (float*)(st);                st += 262144;
  _Float16* hspk0  = (_Float16*)(st);             st += 131072;
  _Float16* hspk1  = (_Float16*)(st);             st += 131072;
  float*    omem   = (float*)(st);                st += 32768;
  float*    ospk   = (float*)(st);                st += 32768;
  _Float16* errh   = (_Float16*)(st);             st += 16384;
  _Float16* errT   = (_Float16*)(st);             st += 16384;
  _Float16* lhT    = (_Float16*)(st);             st += 131072;
  float*    trin   = (float*)(st);                st += 131072;
  _Float16* trinh  = (_Float16*)(st);             st += 65536;
  float*    trrec  = (float*)(st);                st += 262144;
  _Float16* trrech = (_Float16*)(st);             st += 131072;
  float*    trout  = (float*)(st);                st += 262144;
  _Float16* trouth = (_Float16*)(st);             st += 131072;
  float* state_base = hmem;
  int state_floats = (int)(((char*)st - (char*)state_base) / 4);

  // gradient accumulators live in d_out, after outputs [B][T][NOUT]
  float* gin  = dout + (size_t)B_ * T_ * NOUT;       // [NHID][NIN]
  float* grec = gin + (size_t)NHID * NIN;            // [NHID][NHID]
  float* gout = grec + (size_t)NHID * NHID;          // [NOUT][NHID]
  int grad_floats = NHID * NIN + NHID * NHID + NOUT * NHID;

  // ---- prep (idempotent, deterministic) ----
  k_cvt_win <<<(NHID * NIN) / 256, 256, 0, stream>>>(w_in, winh);
  k_wrecT   <<<(NHID * NHID) / 256, 256, 0, stream>>>(w_rec, wrecT);
  k_cvt_wout<<<(NOUT * NHID) / 256, 256, 0, stream>>>(w_out, wouth, woutT);
  k_zero    <<<(state_floats + 255) / 256, 256, 0, stream>>>(state_base, state_floats);
  k_zero    <<<(grad_floats + 255) / 256, 256, 0, stream>>>(gin, grad_floats);

  // ---- T sequential timesteps ----
  for (int t = 0; t < T_; ++t) {
    _Float16* sp_prev = (t & 1) ? hspk1 : hspk0;
    _Float16* sp_next = (t & 1) ? hspk0 : hspk1;
    k_xconv <<<(B_ * NIN) / 256, 256, 0, stream>>>(x, t, xth);
    k_trin  <<<(NIN * B_) / 256, 256, 0, stream>>>(x, t, trin, trinh);
    k_hidden<<<NHID / 32, 128, 0, stream>>>(xth, winh, wrecT, sp_prev, sp_next, hmem);
    k_out   <<<NOUT / 16, 128, 0, stream>>>(sp_next, wouth, label, t, omem, ospk,
                                            dout, errh, errT);
    k_ls    <<<NHID / 16, 128, 0, stream>>>(errh, woutT, hmem, sp_next, lhT,
                                            trrec, trrech, trout, trouth);
    k_grad  <<<1664 / 4, 128, 0, stream>>>(lhT, errT, trinh, trrech, trouth,
                                           gin, grec, gout);
  }
}
